// new_encoder_76501957476756
// MI455X (gfx1250) — matrix operand order repaired and checked by reading, not compiled, not run on hardware
//
#include <hip/hip_runtime.h>
#include <hip/hip_fp16.h>

typedef _Float16 h16;
typedef __attribute__((ext_vector_type(16))) _Float16 v16h;
typedef __attribute__((ext_vector_type(8)))  _Float16 v8h;
typedef __attribute__((ext_vector_type(8)))  float    v8f;

#define BB   1024
#define LL   50
#define EE   20
#define NNb  20
#define MDm  400
#define MDP  416   // MD padded to a multiple of 32 (K-dim for WMMA chains)
#define QDm  200
#define HH   20
#define DKk  20

// ---------------- WMMA helpers (CDNA5 16x16x32 f16 -> f32) ----------------

__device__ __forceinline__ v8f wmma_f16(v16h a, v16h b, v8f c) {
  // (neg_a, A, neg_b, B, c_mod, C, reuse_a, reuse_b)
  return __builtin_amdgcn_wmma_f32_16x16x32_f16(false, a, false, b, (short)0, c, false, false);
}

// A-matrix 16x32 f16 fragment per ISA layout:
// lanes 0-15: row M=lane, halves 0..7 = K kk..kk+7, halves 8..15 = K kk+16..kk+23
// lanes 16-31: row M=lane-16, K ranges shifted by +8
__device__ __forceinline__ v16h load_frag_a(const h16* A, int lda, int m0, int kk, int lane) {
  int row = m0 + (lane & 15);
  int kbase = kk + ((lane & 16) ? 8 : 0);
  const h16* p = A + (size_t)row * lda + kbase;
  v8h lo = *(const v8h*)p;
  v8h hi = *(const v8h*)(p + 16);
  return __builtin_shufflevector(lo, hi, 0,1,2,3,4,5,6,7,8,9,10,11,12,13,14,15);
}

// B-matrix 32x16 f16 fragment, sourced from a transposed ("Bt") [N][K] buffer:
// lane holds column N = n0+(lane&15); lanes 0-15 K kk..kk+7 and kk+16..kk+23, lanes 16-31 shifted by +8
__device__ __forceinline__ v16h load_frag_b(const h16* Bt, int ldb, int n0, int kk, int lane) {
  int col = n0 + (lane & 15);
  int kbase = kk + ((lane & 16) ? 8 : 0);
  const h16* p = Bt + (size_t)col * ldb + kbase;
  v8h lo = *(const v8h*)p;
  v8h hi = *(const v8h*)(p + 16);
  return __builtin_shufflevector(lo, hi, 0,1,2,3,4,5,6,7,8,9,10,11,12,13,14,15);
}

// ---------------- Generic register-blocked WMMA GEMM ----------------
// C[M,N] = act(A[M,Kp] @ Bt[Ncp,Kp]^T + bias); one wave per (16*MB)x(16*NB) tile.
// MB*NB WMMAs share MB+NB fragment loads per K chunk.
// C16 (padded cols zeroed) and/or C32 (only n<N written).
// act: 0 none, 1 tanh, 2 relu, 3 tanh(relu)
template <int MB, int NB>
__global__ __launch_bounds__(32) void
gemm_wmma_t(const h16* __restrict__ A, int lda,
            const h16* __restrict__ Bt, int ldb,
            const float* __restrict__ bias,
            h16* __restrict__ C16, long ldc16,
            float* __restrict__ C32, long ldc32,
            int M, int N, int Kp, int act) {
  int lane = threadIdx.x & 31;
  int n0 = blockIdx.x * (16 * NB);
  int m0 = blockIdx.y * (16 * MB);
  v8f acc[MB][NB];
#pragma unroll
  for (int mi = 0; mi < MB; ++mi)
#pragma unroll
    for (int ni = 0; ni < NB; ++ni) acc[mi][ni] = (v8f){};
  for (int kk = 0; kk < Kp; kk += 32) {
    v16h a[MB], b[NB];
#pragma unroll
    for (int mi = 0; mi < MB; ++mi) a[mi] = load_frag_a(A, lda, m0 + 16 * mi, kk, lane);
#pragma unroll
    for (int ni = 0; ni < NB; ++ni) b[ni] = load_frag_b(Bt, ldb, n0 + 16 * ni, kk, lane);
#pragma unroll
    for (int mi = 0; mi < MB; ++mi)
#pragma unroll
      for (int ni = 0; ni < NB; ++ni)
        acc[mi][ni] = wmma_f16(a[mi], b[ni], acc[mi][ni]);
  }
#pragma unroll
  for (int mi = 0; mi < MB; ++mi) {
    int mb = m0 + 16 * mi + ((lane & 16) ? 8 : 0);
#pragma unroll
    for (int ni = 0; ni < NB; ++ni) {
      int n = n0 + 16 * ni + (lane & 15);
      float bv = (bias != nullptr && n < N) ? bias[n] : 0.f;
#pragma unroll
      for (int i = 0; i < 8; ++i) {
        int m = mb + i;
        float v = acc[mi][ni][i] + bv;
        if (act == 1) v = tanhf(v);
        else if (act == 2) v = fmaxf(v, 0.f);
        else if (act == 3) v = tanhf(fmaxf(v, 0.f));
        if (n >= N) v = 0.f;
        if (C16) C16[(size_t)m * ldc16 + n] = (h16)v;
        if (C32 && n < N) C32[(size_t)m * ldc32 + n] = v;
      }
    }
  }
}

// ---------------- Packing / utility kernels ----------------

__global__ __launch_bounds__(256) void
pack_rows_f16(const float* __restrict__ in, h16* __restrict__ out,
              long M, int K, int Kp) {
  size_t idx = (size_t)blockIdx.x * blockDim.x + threadIdx.x;
  if (idx >= (size_t)M * Kp) return;
  int k = (int)(idx % Kp);
  size_t m = idx / Kp;
  out[idx] = (k < K) ? (h16)in[m * K + k] : (h16)0.f;
}

// W[K,N] (row-major) -> out[Np][Kp] = W^T zero-padded
__global__ __launch_bounds__(256) void
pack_wT_f16(const float* __restrict__ W, h16* __restrict__ out,
            int K, int N, int Kp, int Np) {
  size_t idx = (size_t)blockIdx.x * blockDim.x + threadIdx.x;
  if (idx >= (size_t)Np * Kp) return;
  int k = (int)(idx % Kp);
  int n = (int)(idx / Kp);
  out[idx] = (n < N && k < K) ? (h16)W[(size_t)k * N + n] : (h16)0.f;
}

__global__ __launch_bounds__(256) void
pack_gather_f16(const float* __restrict__ emb, const int* __restrict__ idx,
                h16* __restrict__ out, int K, int Kp) {
  size_t t = (size_t)blockIdx.x * blockDim.x + threadIdx.x;
  if (t >= (size_t)BB * Kp) return;
  int k = (int)(t % Kp);
  int b = (int)(t / Kp);
  out[t] = (k < K) ? (h16)emb[(size_t)idx[b] * K + k] : (h16)0.f;
}

// zero columns [from, stride) of an f16 [rows][stride] buffer
__global__ __launch_bounds__(256) void
zero_pad_cols(h16* __restrict__ buf, long rows, int stride, int from) {
  int w = stride - from;
  size_t idx = (size_t)blockIdx.x * blockDim.x + threadIdx.x;
  if (idx >= (size_t)rows * w) return;
  size_t r = idx / w;
  int c = from + (int)(idx % w);
  buf[r * stride + c] = (h16)0.f;
}

// kg = concat(entity, agg) -> f16 [B*E][224]
__global__ __launch_bounds__(256) void
pack_kg_f16(const float* __restrict__ ent, const float* __restrict__ agg,
            h16* __restrict__ out) {
  size_t idx = (size_t)blockIdx.x * blockDim.x + threadIdx.x;
  if (idx >= (size_t)BB * EE * 224) return;
  int k = (int)(idx % 224);
  size_t r = idx / 224;
  float v = 0.f;
  if (k < 100) v = ent[r * 100 + k];
  else if (k < 200) v = agg[r * 100 + (k - 100)];
  out[idx] = (h16)v;
}

// hbar[b][k] = mean_e he16[b*E+e][k]   (he16 padded cols are zero)
__global__ __launch_bounds__(256) void
mean_he_f16(const h16* __restrict__ he16, h16* __restrict__ hbar) {
  int idx = blockIdx.x * blockDim.x + threadIdx.x;
  if (idx >= BB * 128) return;
  int b = idx / 128, k = idx % 128;
  float s = 0.f;
  for (int e = 0; e < EE; ++e) s += (float)he16[((size_t)b * EE + e) * 128 + k];
  hbar[idx] = (h16)(s * (1.f / EE));
}

// ---------------- Fused KGAT triple-attention ----------------
// per (b,e): att[n] = relu([h|t_n|r_n] @ A1 + a1b) @ A2 ; alpha = softmax(att)
// agg[d] = sum_n alpha[n] * t_n[d].  (a2b cancels in softmax.)
// Register-blocked: per N-tile, one B fragment feeds both M-tiles.
// NOTE: kk loop kept rolled (#pragma unroll 1) so the LDS A-fragment loads stay
// inside the loop — full unroll made LLVM hoist 20 invariant fragments across
// the nt loop and spill them to scratch (observed "Folded Reload" in disasm).
__global__ __launch_bounds__(256) void
kgat_fused(const float* __restrict__ ent,
           const float* __restrict__ nent,
           const float* __restrict__ nrel,
           const h16* __restrict__ A1t,   // [400][320]
           const float* __restrict__ a1b, // [400]
           const float* __restrict__ A2,  // [400]
           float* __restrict__ agg) {     // [B*E][100]
  __shared__ h16  tri[32 * 320];
  __shared__ float attL[32];
  __shared__ float alphaL[32];
  int be = blockIdx.x;
  int b = be / EE, e = be % EE;
  int tid = threadIdx.x;
  for (int i = tid; i < 32 * 320; i += 256) tri[i] = (h16)0.f;
  if (tid < 32) attL[tid] = 0.f;
  __syncthreads();
  const float* entp  = ent  + ((size_t)b * EE + e) * 100;
  const float* nentp = nent + ((size_t)b * EE + e) * NNb * 100;
  const float* nrelp = nrel + ((size_t)b * EE + e) * NNb * 100;
  for (int i = tid; i < NNb * 300; i += 256) {
    int r = i / 300, k = i % 300;
    float v = (k < 100) ? entp[k] : (k < 200) ? nentp[r * 100 + (k - 100)]
                                              : nrelp[r * 100 + (k - 200)];
    tri[r * 320 + k] = (h16)v;
  }
  __syncthreads();
  int wave = tid >> 5, lane = tid & 31;
  for (int nt = wave; nt < 25; nt += 8) {
    int n0 = nt * 16;
    v8f acc0 = {}, acc1 = {};
#pragma unroll 1
    for (int kk = 0; kk < 320; kk += 32) {
      v16h bb = load_frag_b(A1t, 320, n0, kk, lane);
      v16h a0 = load_frag_a(tri, 320, 0, kk, lane);
      v16h a1 = load_frag_a(tri, 320, 16, kk, lane);
      acc0 = wmma_f16(a0, bb, acc0);
      acc1 = wmma_f16(a1, bb, acc1);
    }
    int n = n0 + (lane & 15);
    float a2v = A2[n];
    float bv  = a1b[n];
    int mb = ((lane & 16) ? 8 : 0);
#pragma unroll
    for (int i = 0; i < 8; ++i) {
      int m0i = mb + i;                  // rows 0..15 (< 20, always valid)
      float v0 = fmaxf(acc0[i] + bv, 0.f);
      atomicAdd(&attL[m0i], v0 * a2v);
      int m1 = 16 + mb + i;              // rows 16..31, valid only < 20
      if (m1 < NNb) {
        float v1 = fmaxf(acc1[i] + bv, 0.f);
        atomicAdd(&attL[m1], v1 * a2v);
      }
    }
  }
  __syncthreads();
  if (tid == 0) {
    float mx = -1e30f;
    for (int n = 0; n < NNb; ++n) mx = fmaxf(mx, attL[n]);
    float s = 0.f;
    for (int n = 0; n < NNb; ++n) { float ev = __expf(attL[n] - mx); alphaL[n] = ev; s += ev; }
    float inv = 1.f / s;
    for (int n = 0; n < NNb; ++n) alphaL[n] *= inv;
  }
  __syncthreads();
  for (int d = tid; d < 100; d += 256) {
    float s = 0.f;
    for (int n = 0; n < NNb; ++n) s += alphaL[n] * nentp[n * 100 + d];
    agg[((size_t)b * EE + e) * 100 + d] = s;
  }
}

// ---------------- Fused per-(b,head) MHSA ----------------
__global__ __launch_bounds__(256) void
mhsa_fused(const h16* __restrict__ q16, const h16* __restrict__ k16,
           const h16* __restrict__ v16m, h16* __restrict__ mh16) {
  __shared__ h16  qL[64 * 32];
  __shared__ h16  kL[64 * 32];
  __shared__ h16  vT[32 * 64];
  __shared__ float sL[64 * 64];
  __shared__ h16  pL[64 * 64];
  int bh = blockIdx.x;
  int b = bh / HH, hh = bh % HH;
  int tid = threadIdx.x;
  for (int i = tid; i < 64 * 32; i += 256) { qL[i] = (h16)0.f; kL[i] = (h16)0.f; }
  for (int i = tid; i < 32 * 64; i += 256) vT[i] = (h16)0.f;
  __syncthreads();
  size_t rb = (size_t)b * LL;
  for (int i = tid; i < LL * DKk; i += 256) {
    int l = i / DKk, d = i % DKk;
    size_t src = (rb + l) * MDP + hh * DKk + d;
    qL[l * 32 + d] = q16[src];
    kL[l * 32 + d] = k16[src];
    vT[d * 64 + l] = v16m[src];
  }
  __syncthreads();
  int wave = tid >> 5, lane = tid & 31;
  const float sc = 0.22360679774997896f; // 1/sqrt(DK)
  for (int t = wave; t < 16; t += 8) {   // 4x4 score tiles, K=20->32
    int mt = t >> 2, nt = t & 3;
    v16h a  = load_frag_a(qL, 32, mt * 16, 0, lane);
    v16h bb = load_frag_b(kL, 32, nt * 16, 0, lane);
    v8f acc = {};
    acc = wmma_f16(a, bb, acc);
    int n = nt * 16 + (lane & 15);
    int mb = mt * 16 + ((lane & 16) ? 8 : 0);
#pragma unroll
    for (int i = 0; i < 8; ++i) sL[(size_t)(mb + i) * 64 + n] = acc[i] * sc;
  }
  __syncthreads();
  if (tid < 64) {
    int l = tid;
    if (l < LL) {
      float mx = -1e30f;
      for (int n = 0; n < LL; ++n) mx = fmaxf(mx, sL[l * 64 + n]);
      float s = 0.f;
      for (int n = 0; n < LL; ++n) s += __expf(sL[l * 64 + n] - mx);
      float inv = 1.f / s;
      for (int n = 0; n < 64; ++n)
        pL[l * 64 + n] = (n < LL) ? (h16)(__expf(sL[l * 64 + n] - mx) * inv) : (h16)0.f;
    } else {
      for (int n = 0; n < 64; ++n) pL[l * 64 + n] = (h16)0.f;
    }
  }
  __syncthreads();
  {                                      // P @ v : 4 M-tiles x 2 N-tiles, K=64
    int t = wave;                        // exactly 8 tile pairs
    int mt = t >> 1, nt = t & 1;
    v8f acc = {};
#pragma unroll
    for (int kk = 0; kk < 64; kk += 32) {
      v16h a  = load_frag_a(pL, 64, mt * 16, kk, lane);
      v16h bb = load_frag_b(vT, 64, nt * 16, kk, lane);
      acc = wmma_f16(a, bb, acc);
    }
    int d = nt * 16 + (lane & 15);
    int mb = mt * 16 + ((lane & 16) ? 8 : 0);
#pragma unroll
    for (int i = 0; i < 8; ++i) {
      int l = mb + i;
      if (l < LL && d < DKk)
        mh16[(rb + l) * MDP + hh * DKk + d] = (h16)tanhf(acc[i]);
    }
  }
}

// ---------------- Additive-attention reductions ----------------

__global__ __launch_bounds__(256) void
addattn_word(const h16* __restrict__ T16, const float* __restrict__ qwa,
             const h16* __restrict__ mh16,
             float* __restrict__ reps32, h16* __restrict__ repsh) {
  __shared__ float attL[LL];
  __shared__ float alphaL[LL];
  int b = blockIdx.x, tid = threadIdx.x;
  if (tid < LL) {
    const h16* row = T16 + ((size_t)b * LL + tid) * 208;
    float s = 0.f;
    for (int d = 0; d < QDm; ++d) s += (float)row[d] * qwa[d];
    attL[tid] = s;
  }
  __syncthreads();
  if (tid == 0) {
    float mx = -1e30f;
    for (int l = 0; l < LL; ++l) mx = fmaxf(mx, attL[l]);
    float s = 0.f;
    for (int l = 0; l < LL; ++l) { float ev = __expf(attL[l] - mx); alphaL[l] = ev; s += ev; }
    float inv = 1.f / s;
    for (int l = 0; l < LL; ++l) alphaL[l] *= inv;
  }
  __syncthreads();
  for (int d = tid; d < MDm; d += 256) {
    float s = 0.f;
    for (int l = 0; l < LL; ++l) s += alphaL[l] * (float)mh16[((size_t)b * LL + l) * MDP + d];
    reps32[(size_t)b * 1600 + d] = s;          // view 0 (word)
    repsh [(size_t)b * 4 * MDP + d] = (h16)s;  // row b*4, stride MDP
  }
}

__global__ __launch_bounds__(256) void
final_fuse(const h16* __restrict__ Tf16, const float* __restrict__ qfa,
           const float* __restrict__ reps32, float* __restrict__ out_raw) {
  __shared__ float attL[4];
  __shared__ float alphaL[4];
  int b = blockIdx.x, tid = threadIdx.x;
  if (tid < 4) {
    const h16* row = Tf16 + ((size_t)b * 4 + tid) * 208;
    float s = 0.f;
    for (int d = 0; d < QDm; ++d) s += (float)row[d] * qfa[d];
    attL[tid] = s;
  }
  __syncthreads();
  if (tid == 0) {
    float mx = fmaxf(fmaxf(attL[0], attL[1]), fmaxf(attL[2], attL[3]));
    float s = 0.f;
    for (int v = 0; v < 4; ++v) { float ev = __expf(attL[v] - mx); alphaL[v] = ev; s += ev; }
    float inv = 1.f / s;
    for (int v = 0; v < 4; ++v) alphaL[v] *= inv;
  }
  __syncthreads();
  for (int d = tid; d < MDm; d += 256) {
    float s = 0.f;
    for (int v = 0; v < 4; ++v) s += alphaL[v] * reps32[(size_t)b * 1600 + v * MDm + d];
    out_raw[(size_t)b * MDm + d] = s;
  }
}

__global__ __launch_bounds__(256) void
batchnorm_b(const float* __restrict__ x, const float* __restrict__ gamma,
            const float* __restrict__ beta, float* __restrict__ out) {
  int d = blockIdx.x, tid = threadIdx.x;
  __shared__ float s1[256];
  __shared__ float s2[256];
  float a = 0.f, q = 0.f;
  for (int b = tid; b < BB; b += 256) {
    float v = x[(size_t)b * MDm + d];
    a += v; q += v * v;
  }
  s1[tid] = a; s2[tid] = q;
  __syncthreads();
  for (int st = 128; st > 0; st >>= 1) {
    if (tid < st) { s1[tid] += s1[tid + st]; s2[tid] += s2[tid + st]; }
    __syncthreads();
  }
  float mu  = s1[0] * (1.f / BB);
  float var = s2[0] * (1.f / BB) - mu * mu;
  float inv = rsqrtf(var + 1e-5f) * gamma[d];
  float bt  = beta[d];
  for (int b = tid; b < BB; b += 256) {
    out[(size_t)b * MDm + d] = (x[(size_t)b * MDm + d] - mu) * inv + bt;
  }
}

// ---------------- Host orchestration ----------------

static inline size_t cdiv(size_t a, size_t b) { return (a + b - 1) / b; }

extern "C" void kernel_launch(void* const* d_in, const int* in_sizes, int n_in,
                              void* d_out, int out_size, void* d_ws, size_t ws_size,
                              hipStream_t stream) {
  const float* WE    = (const float*)d_in[0];
  const float* ENT   = (const float*)d_in[1];
  const float* NENT  = (const float*)d_in[2];
  const float* NREL  = (const float*)d_in[3];
  const float* emb_c = (const float*)d_in[4];
  const float* Wc1   = (const float*)d_in[5];
  const float* bc1   = (const float*)d_in[6];
  const float* Wc2   = (const float*)d_in[7];
  const float* bc2   = (const float*)d_in[8];
  const float* emb_s = (const float*)d_in[9];
  const float* Ws1   = (const float*)d_in[10];
  const float* bs1   = (const float*)d_in[11];
  const float* Ws2   = (const float*)d_in[12];
  const float* bs2   = (const float*)d_in[13];
  const float* Ww1   = (const float*)d_in[14];
  const float* bw1   = (const float*)d_in[15];
  const float* Wq    = (const float*)d_in[16];
  const float* bq    = (const float*)d_in[17];
  const float* Wk    = (const float*)d_in[18];
  const float* bk    = (const float*)d_in[19];
  const float* Wv    = (const float*)d_in[20];
  const float* bv    = (const float*)d_in[21];
  const float* Wwa   = (const float*)d_in[22];
  const float* bwa   = (const float*)d_in[23];
  const float* qwa   = (const float*)d_in[24];
  const float* A1    = (const float*)d_in[25];
  const float* a1b   = (const float*)d_in[26];
  const float* A2    = (const float*)d_in[27];
  // d_in[28] a2b: softmax-invariant shift, unused
  const float* We1   = (const float*)d_in[29];
  const float* be1   = (const float*)d_in[30];
  const float* Wg    = (const float*)d_in[31];
  const float* bg    = (const float*)d_in[32];
  // d_in[33..35] Wea/bea/qea: additive attn over identical rows == identity, unused
  const float* Wfa   = (const float*)d_in[36];
  const float* bfa   = (const float*)d_in[37];
  const float* qfa   = (const float*)d_in[38];
  const float* gamma = (const float*)d_in[39];
  const float* beta  = (const float*)d_in[40];
  const int*   cidx  = (const int*)d_in[41];
  const int*   sidx  = (const int*)d_in[42];
  float* out = (float*)d_out;

  char* ws = (char*)d_ws;
  size_t off = 0;
  auto alloc = [&](size_t bytes) -> void* {
    void* p = ws + off;
    off += (bytes + 255) & ~(size_t)255;
    return p;
  };

  h16* WEh   = (h16*)alloc((size_t)51200 * 320 * 2);
  h16* Ww1t  = (h16*)alloc((size_t)128 * 320 * 2);
  h16* h16b  = (h16*)alloc((size_t)51200 * 128 * 2);
  h16* Wqt   = (h16*)alloc((size_t)MDP * 128 * 2);
  h16* Wkt   = (h16*)alloc((size_t)MDP * 128 * 2);
  h16* Wvt   = (h16*)alloc((size_t)MDP * 128 * 2);
  h16* q16   = (h16*)alloc((size_t)51200 * MDP * 2);
  h16* k16   = (h16*)alloc((size_t)51200 * MDP * 2);
  h16* v16b  = (h16*)alloc((size_t)51200 * MDP * 2);
  h16* mh16  = (h16*)alloc((size_t)51200 * MDP * 2);
  h16* Wwat  = (h16*)alloc((size_t)208 * MDP * 2);
  h16* Tw16  = (h16*)alloc((size_t)51200 * 208 * 2);
  h16* A1t   = (h16*)alloc((size_t)400 * 320 * 2);
  float* agg = (float*)alloc((size_t)BB * EE * 100 * 4);
  h16* kg16  = (h16*)alloc((size_t)BB * EE * 224 * 2);
  h16* We1t  = (h16*)alloc((size_t)128 * 224 * 2);
  h16* he16  = (h16*)alloc((size_t)BB * EE * 128 * 2);
  h16* hbar  = (h16*)alloc((size_t)BB * 128 * 2);
  h16* Wgt   = (h16*)alloc((size_t)400 * 128 * 2);
  h16* gc16  = (h16*)alloc((size_t)BB * 128 * 2);
  h16* gs16  = (h16*)alloc((size_t)BB * 128 * 2);
  h16* Wc1t  = (h16*)alloc((size_t)128 * 128 * 2);
  h16* Wc2t  = (h16*)alloc((size_t)400 * 128 * 2);
  h16* Ws1t  = (h16*)alloc((size_t)128 * 128 * 2);
  h16* Ws2t  = (h16*)alloc((size_t)400 * 128 * 2);
  h16* t1c   = (h16*)alloc((size_t)BB * 128 * 2);
  h16* t1s   = (h16*)alloc((size_t)BB * 128 * 2);
  float* reps32 = (float*)alloc((size_t)BB * 1600 * 4);
  h16* repsh = (h16*)alloc((size_t)4096 * MDP * 2);   // [B*4][416] row-major
  h16* Wfat  = (h16*)alloc((size_t)208 * MDP * 2);
  h16* Tf16  = (h16*)alloc((size_t)4096 * 208 * 2);
  float* out_raw = (float*)alloc((size_t)BB * MDm * 4);
  (void)ws_size; (void)n_in; (void)in_sizes; (void)out_size;

  // dispatch: NB==2 requires Ncp % 32 == 0; MB is always 2 (all M % 32 == 0)
  auto gemm = [&](const h16* A, int lda, const h16* Bt, int ldb, const float* bias,
                  h16* C16, long ldc16, float* C32, long ldc32,
                  int M, int N, int Ncp, int Kp, int act, int NB) {
    if (NB == 2) {
      dim3 g(Ncp / 32, M / 32);
      gemm_wmma_t<2, 2><<<g, dim3(32), 0, stream>>>(A, lda, Bt, ldb, bias, C16, ldc16,
                                                    C32, ldc32, M, N, Kp, act);
    } else {
      dim3 g(Ncp / 16, M / 32);
      gemm_wmma_t<2, 1><<<g, dim3(32), 0, stream>>>(A, lda, Bt, ldb, bias, C16, ldc16,
                                                    C32, ldc32, M, N, Kp, act);
    }
  };
  auto packT = [&](const float* W, h16* outp, int K, int N, int Kp, int Np) {
    size_t n = (size_t)Np * Kp;
    pack_wT_f16<<<dim3(cdiv(n, 256)), dim3(256), 0, stream>>>(W, outp, K, N, Kp, Np);
  };

  // ---- zero the pad columns of K-padded chained buffers (written nowhere else) ----
  zero_pad_cols<<<dim3(cdiv((size_t)51200 * 16, 256)), dim3(256), 0, stream>>>(mh16, 51200, MDP, MDm);
  zero_pad_cols<<<dim3(cdiv((size_t)4096 * 16, 256)), dim3(256), 0, stream>>>(repsh, 4096, MDP, MDm);

  // ---- weight packing (tiny, L2-resident afterwards) ----
  packT(Ww1, Ww1t, 300, 100, 320, 128);
  packT(Wq, Wqt, 100, 400, 128, MDP);
  packT(Wk, Wkt, 100, 400, 128, MDP);
  packT(Wv, Wvt, 100, 400, 128, MDP);
  packT(Wwa, Wwat, 400, 200, MDP, 208);
  packT(A1, A1t, 300, 400, 320, 400);
  packT(We1, We1t, 200, 100, 224, 128);
  packT(Wg, Wgt, 100, 400, 128, 400);
  packT(Wc1, Wc1t, 100, 100, 128, 128);
  packT(Wc2, Wc2t, 100, 400, 128, 400);
  packT(Ws1, Ws1t, 100, 100, 128, 128);
  packT(Ws2, Ws2t, 100, 400, 128, 400);
  packT(Wfa, Wfat, 400, 200, MDP, 208);

  // ---- category / subcategory: gather -> GEMM -> GEMM(tanh) into view 2/3 ----
  pack_gather_f16<<<dim3(cdiv((size_t)BB * 128, 256)), dim3(256), 0, stream>>>(emb_c, cidx, gc16, 100, 128);
  pack_gather_f16<<<dim3(cdiv((size_t)BB * 128, 256)), dim3(256), 0, stream>>>(emb_s, sidx, gs16, 100, 128);
  gemm(gc16, 128, Wc1t, 128, bc1, t1c, 128, nullptr, 0, BB, 100, 128, 128, 0, 2);
  gemm(gs16, 128, Ws1t, 128, bs1, t1s, 128, nullptr, 0, BB, 100, 128, 128, 0, 2);
  gemm(t1c, 128, Wc2t, 128, bc2, repsh + 2 * MDP, 4 * MDP, reps32 + 2 * MDm, 1600, BB, 400, 400, 128, 1, 1);
  gemm(t1s, 128, Ws2t, 128, bs2, repsh + 3 * MDP, 4 * MDP, reps32 + 3 * MDm, 1600, BB, 400, 400, 128, 1, 1);

  // ---- word branch ----
  pack_rows_f16<<<dim3(cdiv((size_t)51200 * 320, 256)), dim3(256), 0, stream>>>(WE, WEh, 51200, 300, 320);
  gemm(WEh, 320, Ww1t, 320, bw1, h16b, 128, nullptr, 0, 51200, 100, 128, 320, 1, 2);  // h=tanh(..)
  gemm(h16b, 128, Wqt, 128, bq, q16, MDP, nullptr, 0, 51200, 400, MDP, 128, 0, 2);
  gemm(h16b, 128, Wkt, 128, bk, k16, MDP, nullptr, 0, 51200, 400, MDP, 128, 0, 2);
  gemm(h16b, 128, Wvt, 128, bv, v16b, MDP, nullptr, 0, 51200, 400, MDP, 128, 0, 2);
  mhsa_fused<<<dim3(BB * HH), dim3(256), 0, stream>>>(q16, k16, v16b, mh16);
  gemm(mh16, MDP, Wwat, MDP, bwa, Tw16, 208, nullptr, 0, 51200, 200, 208, MDP, 1, 1); // tanh
  addattn_word<<<dim3(BB), dim3(256), 0, stream>>>(Tw16, qwa, mh16, reps32, repsh);   // view 0

  // ---- KGAT + entity branch ----
  kgat_fused<<<dim3(BB * EE), dim3(256), 0, stream>>>(ENT, NENT, NREL, A1t, a1b, A2, agg);
  pack_kg_f16<<<dim3(cdiv((size_t)BB * EE * 224, 256)), dim3(256), 0, stream>>>(ENT, agg, kg16);
  gemm(kg16, 224, We1t, 224, be1, he16, 128, nullptr, 0, BB * EE, 100, 128, 224, 1, 2); // tanh
  mean_he_f16<<<dim3(cdiv((size_t)BB * 128, 256)), dim3(256), 0, stream>>>(he16, hbar);
  // gcn rows are identical across E -> additive attn is identity: rep = tanh(relu(.))
  gemm(hbar, 128, Wgt, 128, bg, repsh + 1 * MDP, 4 * MDP, reps32 + 1 * MDm, 1600, BB, 400, 400, 128, 3, 1);

  // ---- final fuse + batchnorm ----
  gemm(repsh, MDP, Wfat, MDP, bfa, Tf16, 208, nullptr, 0, 4096, 200, 208, MDP, 1, 1); // tanh
  final_fuse<<<dim3(BB), dim3(256), 0, stream>>>(Tf16, qfa, reps32, out_raw);
  batchnorm_b<<<dim3(MDm), dim3(256), 0, stream>>>(out_raw, gamma, beta, out);
}
